// DenseSAKELayer_20564303413685
// MI455X (gfx1250) — compile-verified, matrix operand order checked by reading, not run on hardware
//
#include <hip/hip_runtime.h>

typedef __attribute__((ext_vector_type(4)))  _Float16 v4h;
typedef __attribute__((ext_vector_type(8)))  _Float16 v8h;
typedef __attribute__((ext_vector_type(16))) _Float16 v16h;
typedef __attribute__((ext_vector_type(8)))  float    v8f;

#define NB   2
#define NN   256
#define FF   64
#define NH   4
#define HID  64
#define CC   256
#define KF   50
#define ASTR 264                 // padded half-stride of LDS A-matrix rows (528 B, 16B aligned)
#define USTR 72                  // half-stride for U/T/E phase-A scratch (144 B, 16B aligned)

// LDS layout (bytes)
#define LDS_A_BYTES   (256 * ASTR * 2)       // 135168 (also hosts U/T/E during phase A)
#define LDS_XHAT_OFF  (LDS_A_BYTES)          // 256*4 floats (xhat.xyz, d)
#define LDS_COMB_OFF  (LDS_XHAT_OFF + 4096)  // 256*4 floats
#define LDS_RED_OFF   (LDS_COMB_OFF + 4096)  // 1024 floats
#define LDS_HE_OFF    (LDS_RED_OFF + 4096)   // 256 floats
#define LDS_MISC_OFF  (LDS_HE_OFF + 1024)    // 384 floats
#define LDS_BYTES     (LDS_MISC_OFF + 1536)

// phase-A scratch offsets inside the A region (halves)
#define U_OFF 0
#define T_OFF (256 * USTR)
#define E_OFF (2 * 256 * USTR)                // ends at 110592 B < 135168 B

// lds_misc float slots
#define MQ    0     // Q[i] 50
#define MRI   64    // Ri[i] 64
#define MHR   128   // h row i 64
#define MS1   192   // mlp scratch 64
#define MB2   256   // mlp scratch 64
#define MXI   320   // x_i 4
#define MSC   324   // vscale

__device__ __forceinline__ float frcp(float x)  { return __builtin_amdgcn_rcpf(x); }
__device__ __forceinline__ float fsilu(float x) { return x * frcp(1.f + __expf(-x)); }
__device__ __forceinline__ float fsig(float x)  { return frcp(1.f + __expf(-x)); }
__device__ __forceinline__ float ftanh(float x) {
#if __has_builtin(__builtin_amdgcn_tanhf)
    return __builtin_amdgcn_tanhf(x);                 // native V_TANH_F32 on gfx1250
#else
    float r;
    asm("v_tanh_f32 %0, %1" : "=v"(r) : "v"(x));      // CDNA5 transcendental (ISA §8.4)
    return r;
#endif
}
__device__ __forceinline__ float fcelu2(float x){ return x > 0.f ? x : 2.f * (__expf(0.5f * x) - 1.f); }

__device__ __forceinline__ v16h cat8(v8h lo, v8h hi) {
    return __builtin_shufflevector(lo, hi, 0,1,2,3,4,5,6,7,8,9,10,11,12,13,14,15);
}

// ---------------------------------------------------------------------------
// Prep 1: per-node factorizations of the pair GEMMs.
//   P = h @ W_in[0:64] + b_in ; Q = h @ W_in[64:128]
//   RJ = h @ W_o1[0:64]       ; RI = h @ W_o1[64:128]
// ---------------------------------------------------------------------------
__global__ void sake_prep_pq(const float* __restrict__ h, const float* __restrict__ W_in,
                             const float* __restrict__ b_in, const float* __restrict__ W_o1,
                             float* __restrict__ P, float* __restrict__ Q,
                             float* __restrict__ RJ, float* __restrict__ RI)
{
    __shared__ float hrow[FF];
    const int bn = blockIdx.x;
    const int t  = threadIdx.x;     // 64
    hrow[t] = h[bn * FF + t];
    __syncthreads();

    float aj = 0.f, ai = 0.f;
    for (int f = 0; f < FF; ++f) {
        const float hv = hrow[f];
        aj += hv * W_o1[f * HID + t];
        ai += hv * W_o1[(FF + f) * HID + t];
    }
    RJ[bn * HID + t] = aj;
    RI[bn * HID + t] = ai;

    if (t < KF) {
        float p = b_in[t], q = 0.f;
        for (int f = 0; f < FF; ++f) {
            const float hv = hrow[f];
            p += hv * W_in[f * KF + t];
            q += hv * W_in[(FF + f) * KF + t];
        }
        P[bn * KF + t] = p;
        Q[bn * KF + t] = q;
    }
}

// ---------------------------------------------------------------------------
// Prep 2: swizzle a [rows x N] f32 matrix (leading dim srcLd) into the f16
// WMMA B-operand lane layout, zero-padding K beyond `rows`.
// blockDim.x = ktCount*32, gridDim.x = N/16.
// ---------------------------------------------------------------------------
__global__ void sake_prep_swz(const float* __restrict__ src, int srcLd, int rows,
                              _Float16* __restrict__ dst)
{
    const int ct    = blockIdx.x;
    const int ktCnt = blockDim.x >> 5;
    const int kt    = threadIdx.x >> 5;
    const int lane  = threadIdx.x & 31;
    const int group = lane >> 4;
    const int n     = ct * 16 + (lane & 15);
    _Float16* d = dst + ((size_t)(ct * ktCnt + kt) * 32 + lane) * 16;
    for (int hh = 0; hh < 8; ++hh) {
        const int k0 = kt * 32 + group * 8 + hh;
        const int k1 = k0 + 16;
        d[hh]     = (_Float16)((k0 < rows) ? src[(size_t)k0 * srcLd + n] : 0.f);
        d[8 + hh] = (_Float16)((k1 < rows) ? src[(size_t)k1 * srcLd + n] : 0.f);
    }
}

// ---------------------------------------------------------------------------
// Fused main kernel: one workgroup per (b, i); 256 threads = 8 wave32.
// ---------------------------------------------------------------------------
__global__ void __launch_bounds__(256) sake_main(
    const float* __restrict__ h, const float* __restrict__ x, const float* __restrict__ v,
    const float* __restrict__ rbf_means, const float* __restrict__ rbf_betas,
    const float* __restrict__ W_o1, const float* __restrict__ b_o1,
    const float* __restrict__ b_o2,
    const float* __restrict__ W_sem, const float* __restrict__ b_sem,
    const float* __restrict__ W_p1, const float* __restrict__ b_p1,
    const float* __restrict__ W_p2, const float* __restrict__ b_p2,
    const float* __restrict__ W_n1, const float* __restrict__ b_n1,
    const float* __restrict__ W_n2, const float* __restrict__ b_n2,
    const float* __restrict__ W_v1, const float* __restrict__ b_v1,
    const float* __restrict__ W_v2, const float* __restrict__ W_vmix,
    const float* __restrict__ P, const float* __restrict__ Q,
    const float* __restrict__ RJ, const float* __restrict__ RI,
    const _Float16* __restrict__ WXS,   // swizzled W_xmix  [16ct x 8kt]
    const _Float16* __restrict__ WXA,   // swizzled W_o1c   [4ct x 2kt] (K padded 50->64)
    const _Float16* __restrict__ WXB,   // swizzled W_o2    [4ct x 2kt]
    float* __restrict__ out_h, float* __restrict__ out_x, float* __restrict__ out_v)
{
    extern __shared__ char smem[];
    _Float16* lds_A    = (_Float16*)smem;
    _Float16* ldsU     = lds_A + U_OFF;
    _Float16* ldsT     = lds_A + T_OFF;
    _Float16* ldsE     = lds_A + E_OFF;
    float*    lds_xhat = (float*)(smem + LDS_XHAT_OFF);
    float*    lds_comb = (float*)(smem + LDS_COMB_OFF);
    float*    lds_red  = (float*)(smem + LDS_RED_OFF);
    float*    lds_he   = (float*)(smem + LDS_HE_OFF);
    float*    lds_misc = (float*)(smem + LDS_MISC_OFF);

    const int bi = blockIdx.x;           // b*256 + i
    const int b  = bi >> 8;
    const int i  = bi & 255;
    const int j  = threadIdx.x;
    const int bj = b * NN + j;
    const int wave = j >> 5, lane = j & 31;
    const int group = lane >> 4, mrow = lane & 15;

    // ---- Phase 0: stage per-i data ----
    if (j < KF)  lds_misc[MQ  + j] = Q[bi * KF + j];
    if (j < HID) lds_misc[MRI + j] = RI[bi * HID + j];
    if (j < FF)  lds_misc[MHR + j] = h[bi * FF + j];
    if (j < 4)   lds_misc[MXI + j] = (j < 3) ? x[bi * 3 + j] : 0.f;
    lds_he[j] = 0.f;                     // accumulated atomically in phase C
    __syncthreads();

    // ---- Phase A1: geometry + RBF-filtered features U (per pair row j) ----
    const float xi0 = lds_misc[MXI + 0], xi1 = lds_misc[MXI + 1], xi2 = lds_misc[MXI + 2];
    const float dx = x[bj * 3 + 0] - xi0;
    const float dy = x[bj * 3 + 1] - xi1;
    const float dz = x[bj * 3 + 2] - xi2;
    const float d = sqrtf(fmaxf(dx * dx + dy * dy + dz * dz, 0.f) + 1e-5f);
    const float invd = frcp(d + 1e-5f);
    lds_xhat[j * 4 + 0] = dx * invd;
    lds_xhat[j * 4 + 1] = dy * invd;
    lds_xhat[j * 4 + 2] = dz * invd;
    lds_xhat[j * 4 + 3] = d;                       // stashed for GEMM1 epilogue

    __builtin_prefetch(RJ + (size_t)bj * HID, 0, 3);

    {   // zero-pad U row, then fill k = 0..49
        v8h z = {};
        #pragma unroll
        for (int p = 0; p < 8; ++p) *(v8h*)(ldsU + (size_t)j * USTR + p * 8) = z;
        const float emd  = __expf(-d);             // ALPHA = 1
        const float cutf = (d < 5.0f) ? 0.5f * (__cosf(d * 0.6283185307179586f) + 1.f) : 0.f;
        const float* prow = P + (size_t)bj * KF;
        for (int k = 0; k < KF; ++k) {
            const float h1 = prow[k] + lds_misc[MQ + k];
            const float dm = emd - rbf_means[k];
            const float uk = cutf * __expf(-rbf_betas[k] * dm * dm) * h1;
            ldsU[(size_t)j * USTR + k] = (_Float16)uk;
        }
    }
    __syncthreads();

    // ---- GEMM1 (WMMA): T = silu(U @ W_o1c + Rj + Ri + d*w_d + b_o1) ----
    {
        v16h B1[8];
        #pragma unroll
        for (int tt = 0; tt < 8; ++tt) {
            const _Float16* src = WXA + ((size_t)tt * 32 + lane) * 16;
            B1[tt] = cat8(*(const v8h*)src, *(const v8h*)(src + 8));
        }
        for (int q = 0; q < 2; ++q) {
            const int jt = wave * 2 + q;
            #pragma unroll
            for (int ct = 0; ct < 4; ++ct) {
                v8f acc = {};
                #pragma unroll
                for (int kt = 0; kt < 2; ++kt) {
                    const _Float16* ap = ldsU + (size_t)(jt * 16 + mrow) * USTR + kt * 32 + group * 8;
                    acc = __builtin_amdgcn_wmma_f32_16x16x32_f16(
                        false, cat8(*(const v8h*)ap, *(const v8h*)(ap + 16)),
                        false, B1[ct * 2 + kt], (short)0, acc, false, false);
                }
                const int n = ct * 16 + mrow;
                const float ribn = lds_misc[MRI + n] + b_o1[n];
                const float wdn  = W_o1[178 * HID + n];
                const int mbase  = jt * 16 + group * 8;
                #pragma unroll
                for (int r = 0; r < 8; ++r) {
                    const int m = mbase + r;
                    const float z = acc[r] + RJ[(size_t)(b * NN + m) * HID + n]
                                  + ribn + lds_xhat[m * 4 + 3] * wdn;
                    ldsT[(size_t)m * USTR + n] = (_Float16)fsilu(z);
                }
            }
        }
    }
    __syncthreads();

    // ---- GEMM2 (WMMA): E = T @ W_o2 + b_o2 ----
    {
        v16h B2[8];
        #pragma unroll
        for (int tt = 0; tt < 8; ++tt) {
            const _Float16* src = WXB + ((size_t)tt * 32 + lane) * 16;
            B2[tt] = cat8(*(const v8h*)src, *(const v8h*)(src + 8));
        }
        for (int q = 0; q < 2; ++q) {
            const int jt = wave * 2 + q;
            #pragma unroll
            for (int ct = 0; ct < 4; ++ct) {
                v8f acc = {};
                #pragma unroll
                for (int kt = 0; kt < 2; ++kt) {
                    const _Float16* ap = ldsT + (size_t)(jt * 16 + mrow) * USTR + kt * 32 + group * 8;
                    acc = __builtin_amdgcn_wmma_f32_16x16x32_f16(
                        false, cat8(*(const v8h*)ap, *(const v8h*)(ap + 16)),
                        false, B2[ct * 2 + kt], (short)0, acc, false, false);
                }
                const int n = ct * 16 + mrow;
                const float bo2 = b_o2[n];
                const int mbase = jt * 16 + group * 8;
                #pragma unroll
                for (int r = 0; r < 8; ++r)
                    ldsE[(size_t)(mbase + r) * USTR + n] = (_Float16)(acc[r] + bo2);
            }
        }
    }
    __syncthreads();

    // ---- Phase A2: attention logits + softmax over j ----
    v8h er8[8];                              // this thread's E row (64 f16 in 32 VGPRs)
    {
        const _Float16* er = ldsE + (size_t)j * USTR;
        #pragma unroll
        for (int p = 0; p < 8; ++p) er8[p] = *(const v8h*)(er + p * 8);
    }
    float a0 = b_sem[0], a1 = b_sem[1], a2 = b_sem[2], a3 = b_sem[3];
    #pragma unroll
    for (int p = 0; p < 8; ++p) {
        #pragma unroll
        for (int qq = 0; qq < 8; ++qq) {
            const float ev = (float)er8[p][qq];
            const float* ws = W_sem + (p * 8 + qq) * NH;
            a0 += ev * ws[0]; a1 += ev * ws[1]; a2 += ev * ws[2]; a3 += ev * ws[3];
        }
    }
    a0 = fcelu2(a0); a1 = fcelu2(a1); a2 = fcelu2(a2); a3 = fcelu2(a3);
    if (j == i) { a0 -= 1e5f; a1 -= 1e5f; a2 -= 1e5f; a3 -= 1e5f; }

    lds_red[0 * NN + j] = a0; lds_red[1 * NN + j] = a1;
    lds_red[2 * NN + j] = a2; lds_red[3 * NN + j] = a3;
    __syncthreads();
    for (int s = 128; s > 0; s >>= 1) {
        if (j < s) {
            #pragma unroll
            for (int hh = 0; hh < NH; ++hh)
                lds_red[hh * NN + j] = fmaxf(lds_red[hh * NN + j], lds_red[hh * NN + j + s]);
        }
        __syncthreads();
    }
    const float m0 = lds_red[0], m1 = lds_red[NN], m2 = lds_red[2 * NN], m3 = lds_red[3 * NN];
    __syncthreads();
    const float ex0 = __expf(a0 - m0), ex1 = __expf(a1 - m1);
    const float ex2 = __expf(a2 - m2), ex3 = __expf(a3 - m3);
    lds_red[0 * NN + j] = ex0; lds_red[1 * NN + j] = ex1;
    lds_red[2 * NN + j] = ex2; lds_red[3 * NN + j] = ex3;
    __syncthreads();
    for (int s = 128; s > 0; s >>= 1) {
        if (j < s) {
            #pragma unroll
            for (int hh = 0; hh < NH; ++hh)
                lds_red[hh * NN + j] += lds_red[hh * NN + j + s];
        }
        __syncthreads();
    }
    const float c0 = ex0 * frcp(lds_red[0]),      c1 = ex1 * frcp(lds_red[NN]);
    const float c2 = ex2 * frcp(lds_red[2 * NN]), c3 = ex3 * frcp(lds_red[3 * NN]);
    __syncthreads();                          // all E reads done; safe to overwrite region

    // h_e_att row j -> LDS f16 (c = hid*4 + head); overwrites U/T/E scratch
    #pragma unroll
    for (int p = 0; p < 8; ++p) {
        #pragma unroll
        for (int qq = 0; qq < 8; qq += 2) {
            const float e0 = (float)er8[p][qq], e1 = (float)er8[p][qq + 1];
            v8h pk = { (_Float16)(e0 * c0), (_Float16)(e0 * c1),
                       (_Float16)(e0 * c2), (_Float16)(e0 * c3),
                       (_Float16)(e1 * c0), (_Float16)(e1 * c1),
                       (_Float16)(e1 * c2), (_Float16)(e1 * c3) };
            *(v8h*)(lds_A + (size_t)j * ASTR + (p * 8 + qq) * 4) = pk;
        }
    }
    __syncthreads();

    // ---- Phase B (WMMA): coeff = tanh(A @ W_xmix), fused comb_sum reduction ----
    for (int cti = 0; cti < 2; ++cti) {
        const int ct = wave * 2 + cti;
        v16h Bt[8];
        #pragma unroll
        for (int kt = 0; kt < 8; ++kt) {
            const _Float16* src = WXS + ((size_t)(ct * 8 + kt) * 32 + lane) * 16;
            Bt[kt] = cat8(*(const v8h*)src, *(const v8h*)(src + 8));
        }
        float ac0 = 0.f, ac1 = 0.f, ac2 = 0.f;   // lane column c = ct*16+mrow
        for (int jt = 0; jt < 16; ++jt) {
            v8f acc = {};
            #pragma unroll
            for (int kt = 0; kt < 8; ++kt) {
                const _Float16* ap = lds_A + (size_t)(jt * 16 + mrow) * ASTR + kt * 32 + group * 8;
                acc = __builtin_amdgcn_wmma_f32_16x16x32_f16(
                    false, cat8(*(const v8h*)ap, *(const v8h*)(ap + 16)),
                    false, Bt[kt], (short)0, acc, false, false);
            }
            const int mbase = jt * 16 + group * 8;
            #pragma unroll
            for (int r = 0; r < 8; ++r) {
                const float cf = ftanh(acc[r]);
                const float* xh = lds_xhat + (mbase + r) * 4;
                ac0 += cf * xh[0]; ac1 += cf * xh[1]; ac2 += cf * xh[2];
            }
        }
        ac0 += __shfl_xor(ac0, 16, 32);
        ac1 += __shfl_xor(ac1, 16, 32);
        ac2 += __shfl_xor(ac2, 16, 32);
        if (group == 0) {
            const int c = ct * 16 + mrow;
            lds_comb[c * 4 + 0] = ac0 * (1.f / 256.f);
            lds_comb[c * 4 + 1] = ac1 * (1.f / 256.f);
            lds_comb[c * 4 + 2] = ac2 * (1.f / 256.f);
        }
    }
    __syncthreads();

    // ---- Phase C: reductions + small MLPs ----
    {   // h_e column sums: 32-row x 8-col block per thread, b128 LDS loads
        const int cb = j & 31;               // column block (8 cols)
        const int rg = j >> 5;               // row group (32 rows)
        float acc8[8] = {0.f, 0.f, 0.f, 0.f, 0.f, 0.f, 0.f, 0.f};
        for (int r = 0; r < 32; ++r) {
            v8h ch = *(const v8h*)(lds_A + (size_t)(rg * 32 + r) * ASTR + cb * 8);
            #pragma unroll
            for (int q = 0; q < 8; ++q) acc8[q] += (float)ch[q];
        }
        #pragma unroll
        for (int q = 0; q < 8; ++q) atomicAdd(&lds_he[cb * 8 + q], acc8[q]);
    }
    {   // comb_norm
        const float q0 = lds_comb[j * 4 + 0], q1 = lds_comb[j * 4 + 1], q2 = lds_comb[j * 4 + 2];
        lds_red[j] = q0 * q0 + q1 * q1 + q2 * q2;
    }
    __syncthreads();
    if (j < HID) {                                   // post_norm_mlp stage 1
        float acc = b_p1[j];
        for (int cc = 0; cc < CC; ++cc) acc += lds_red[cc] * W_p1[cc * HID + j];
        lds_misc[MS1 + j] = fsilu(acc);
    }
    __syncthreads();
    if (j < HID) {                                   // post_norm_mlp stage 2 -> h_comb
        float acc = b_p2[j];
        for (int k = 0; k < HID; ++k) acc += lds_misc[MS1 + k] * W_p2[k * HID + j];
        lds_misc[MB2 + j] = fsilu(acc);
    }
    __syncthreads();
    if (j < HID) {                                   // node mlp stage 1
        float acc = b_n1[j];
        for (int f = 0; f < FF; ++f)    acc += lds_misc[MHR + f] * W_n1[f * HID + j];
        for (int cc = 0; cc < CC; ++cc) acc += lds_he[cc] * W_n1[(FF + cc) * HID + j];
        for (int f = 0; f < HID; ++f)   acc += lds_misc[MB2 + f] * W_n1[(FF + CC + f) * HID + j];
        lds_misc[MS1 + j] = fsilu(acc);
    }
    __syncthreads();
    if (j < HID) {                                   // node mlp stage 2 + residual
        float acc = b_n2[j];
        for (int k = 0; k < HID; ++k) acc += lds_misc[MS1 + k] * W_n2[k * HID + j];
        const float hn = lds_misc[MHR + j] + fsilu(acc);
        lds_misc[MB2 + j] = hn;                      // reuse as h_new
        out_h[(size_t)bi * HID + j] = hn;
    }
    __syncthreads();
    if (j < HID) {                                   // velocity mlp stage 1
        float acc = b_v1[j];
        for (int f = 0; f < HID; ++f) acc += lds_misc[MB2 + f] * W_v1[f * HID + j];
        lds_red[j] = fsilu(acc) * W_v2[j];
    }
    __syncthreads();
    for (int s = 32; s > 0; s >>= 1) {               // reduce 64 -> v_scale
        if (j < s) lds_red[j] += lds_red[j + s];
        __syncthreads();
    }
    if (j == 0) lds_misc[MSC] = 2.f * fsig(lds_red[0]);
    __syncthreads();
    {   // delta_v[d] = sum_c comb[c][d]*W_vmix[c] : one 3-wide tree
        const float wv = W_vmix[j];
        lds_red[0 * NN + j] = lds_comb[j * 4 + 0] * wv;
        lds_red[1 * NN + j] = lds_comb[j * 4 + 1] * wv;
        lds_red[2 * NN + j] = lds_comb[j * 4 + 2] * wv;
    }
    __syncthreads();
    for (int s = 128; s > 0; s >>= 1) {
        if (j < s) {
            lds_red[0 * NN + j] += lds_red[0 * NN + j + s];
            lds_red[1 * NN + j] += lds_red[1 * NN + j + s];
            lds_red[2 * NN + j] += lds_red[2 * NN + j + s];
        }
        __syncthreads();
    }
    if (j < 3) {
        const float vs = lds_misc[MSC];
        const float vn = lds_red[j * NN] + vs * v[bi * 3 + j];
        out_v[(size_t)bi * 3 + j] = vn;
        out_x[(size_t)bi * 3 + j] = x[bi * 3 + j] + vn;
    }
}

extern "C" void kernel_launch(void* const* d_in, const int* in_sizes, int n_in,
                              void* d_out, int out_size, void* d_ws, size_t ws_size,
                              hipStream_t stream)
{
    (void)in_sizes; (void)n_in; (void)out_size; (void)ws_size;
    const float* h         = (const float*)d_in[0];
    const float* x         = (const float*)d_in[1];
    const float* v         = (const float*)d_in[2];
    const float* W_in      = (const float*)d_in[3];
    const float* b_in      = (const float*)d_in[4];
    const float* rbf_means = (const float*)d_in[5];
    const float* rbf_betas = (const float*)d_in[6];
    const float* W_o1 = (const float*)d_in[7];  const float* b_o1 = (const float*)d_in[8];
    const float* W_o2 = (const float*)d_in[9];  const float* b_o2 = (const float*)d_in[10];
    const float* W_sem = (const float*)d_in[11]; const float* b_sem = (const float*)d_in[12];
    const float* W_xmix = (const float*)d_in[13];
    const float* W_p1 = (const float*)d_in[14]; const float* b_p1 = (const float*)d_in[15];
    const float* W_p2 = (const float*)d_in[16]; const float* b_p2 = (const float*)d_in[17];
    const float* W_n1 = (const float*)d_in[18]; const float* b_n1 = (const float*)d_in[19];
    const float* W_n2 = (const float*)d_in[20]; const float* b_n2 = (const float*)d_in[21];
    const float* W_v1 = (const float*)d_in[22]; const float* b_v1 = (const float*)d_in[23];
    const float* W_v2 = (const float*)d_in[24]; const float* W_vmix = (const float*)d_in[25];

    // workspace carve-out
    float* P  = (float*)d_ws;            // [B*N*50]
    float* Q  = P  + NB * NN * KF;       // [B*N*50]
    float* RJ = Q  + NB * NN * KF;       // [B*N*64]
    float* RI = RJ + NB * NN * HID;      // [B*N*64]
    _Float16* WXS = (_Float16*)(RI + NB * NN * HID);  // 65536 halves (swizzled W_xmix)
    _Float16* WXA = WXS + 65536;                      // 4096 halves (swizzled W_o1c)
    _Float16* WXB = WXA + 4096;                       // 4096 halves (swizzled W_o2)

    float* out_h = (float*)d_out;                    // [2,256,64]
    float* out_x = out_h + NB * NN * HID;            // [2,256,3]
    float* out_v = out_x + NB * NN * 3;              // [2,256,3]

    sake_prep_pq<<<NB * NN, FF, 0, stream>>>(h, W_in, b_in, W_o1, P, Q, RJ, RI);
    sake_prep_swz<<<16, 256, 0, stream>>>(W_xmix, CC, CC, WXS);           // 16ct x 8kt
    sake_prep_swz<<<4, 64, 0, stream>>>(W_o1 + 128 * HID, HID, KF, WXA);  // 4ct x 2kt, pad 50->64
    sake_prep_swz<<<4, 64, 0, stream>>>(W_o2, HID, HID, WXB);             // 4ct x 2kt
    sake_main<<<NB * NN, 256, LDS_BYTES, stream>>>(
        h, x, v, rbf_means, rbf_betas,
        W_o1, b_o1, b_o2, W_sem, b_sem,
        W_p1, b_p1, W_p2, b_p2, W_n1, b_n1, W_n2, b_n2,
        W_v1, b_v1, W_v2, W_vmix,
        P, Q, RJ, RI, WXS, WXA, WXB, out_h, out_x, out_v);
}
